// DCRNN_7593502179662
// MI455X (gfx1250) — matrix operand order repaired and checked by reading, not compiled, not run on hardware
//
#include <hip/hip_runtime.h>
#include <hip/hip_bf16.h>

// ---------------- problem constants ----------------
#define NN   4096
#define BB   32
#define TT   12
#define FIN  2
#define HH   64
#define KDIF 2      // number of diffusion supports
#define HOR  12
#define KPAD 96     // padded GRU GEMM reduction dim (enc: 68 used, dec: 66 used)
#define SPLITK 4    // diffusion split-K factor (occupancy: 32*4*2 = 256 WGs)

typedef __attribute__((ext_vector_type(16))) __bf16 v16bf;
typedef __attribute__((ext_vector_type(8)))  float  v8f;
typedef unsigned short u16;
typedef unsigned int   u32;

struct alignas(16) U4 { u32 x, y, z, w; };

union Frag {
    v16bf v;
    u16   u[16];
    U4    q[2];
};

__device__ __forceinline__ u16 f2bf(float x) {
    union { float f; u32 u; } c; c.f = x;
    u32 u = c.u;
    u32 r = (u + 0x7FFFu + ((u >> 16) & 1u)) >> 16;  // round-to-nearest-even
    return (u16)r;
}

__device__ __forceinline__ v8f wmma_bf16(Frag a, Frag b, v8f c) {
    // D = A(16x32 bf16) * B(32x16 bf16) + C(16x16 f32)
    return __builtin_amdgcn_wmma_f32_16x16x32_bf16(
        /*neg_a=*/false, a.v, /*neg_b=*/false, b.v,
        /*c_mod=*/(short)0, c, /*reuse_a=*/false, /*reuse_b=*/false);
}

// ---------------- prep kernels ----------------
__global__ void bf16cast_kernel(const float* __restrict__ in, u16* __restrict__ out, size_t n) {
    size_t i = (size_t)blockIdx.x * blockDim.x + threadIdx.x;
    size_t stride = (size_t)gridDim.x * blockDim.x;
    for (; i < n; i += stride) out[i] = f2bf(in[i]);
}

__global__ void zero_kernel(u32* __restrict__ p, size_t nWords) {
    size_t i = (size_t)blockIdx.x * blockDim.x + threadIdx.x;
    size_t stride = (size_t)gridDim.x * blockDim.x;
    for (; i < nWords; i += stride) p[i] = 0u;
}

// Pack W ([inDim][64] row-major, optionally two concatenated along cols) into
// column-major, K-padded bf16: out[c][k], c in [0,ncols), k in [0,KPAD)
__global__ void pack_kernel(const float* __restrict__ W1, const float* __restrict__ W2,
                            u16* __restrict__ out, int inDim, int ncols) {
    int i = blockIdx.x * blockDim.x + threadIdx.x;
    int total = ncols * KPAD;
    if (i >= total) return;
    int c = i / KPAD, k = i % KPAD;
    float v = 0.f;
    if (k < inDim) v = (c < HH) ? W1[k * HH + c] : W2[k * HH + (c - HH)];
    out[i] = f2bf(v);
}

// Pack the diffusion B panel to bf16, column-major: xpk[c][k], c = b*F+fi, k = node.
__global__ void xpack_kernel(const float* __restrict__ xsrc, long sB, long sN, int F,
                             u16* __restrict__ xpk, int cols) {
    int total = cols * NN;
    int i = blockIdx.x * blockDim.x + threadIdx.x;
    if (i >= total) return;
    int c = i / NN, k = i % NN;
    xpk[i] = f2bf(xsrc[(size_t)(c / F) * sB + (size_t)k * sN + (c % F)]);
}

// Merge split-K partials -> bf16 x_cat columns of both activation matrices.
__global__ void xmerge_kernel(const float* __restrict__ xpart,  // [SPLITK][ROWS][4]
                              u16* __restrict__ A1, u16* __restrict__ A2, int KF) {
    const size_t ROWS = (size_t)BB * NN;
    size_t total = ROWS * KF;
    size_t i = (size_t)blockIdx.x * blockDim.x + threadIdx.x;
    if (i >= total) return;
    size_t row = i / KF;
    int c = (int)(i % KF);
    float s = 0.f;
#pragma unroll
    for (int k = 0; k < SPLITK; ++k) s += xpart[(k * ROWS + row) * 4 + c];
    u16 hv = f2bf(s);
    A1[row * KPAD + c] = hv;
    A2[row * KPAD + c] = hv;
}

// ---------------- diffusion GEMM: x_cat partial = supports[sup][kchunk] @ x ----------------
// C = A(4096x4096 bf16) * X(4096 x BB*F), per-support (blockIdx.z), split-K (blockIdx.y).
// Workgroup: 8 waves, each computes a 16-row x (BB*F)-col slab; M-tile = 128.
// Pure global_load_b128 -> v_wmma pipeline (x panel pre-packed bf16 col-major, L2-hot).
template <int F>
__global__ __launch_bounds__(256) void diffusion_kernel(
    const u16* __restrict__ supB,      // [KDIF][NN][NN] bf16 row-major
    const u16* __restrict__ xpk,       // [BB*F][NN] bf16 col-major panel
    float* __restrict__ xpart)         // [SPLITK][BB*NN][4] f32 partials
{
    constexpr int COLS = BB * F;       // 64 (enc) or 32 (dec)
    constexpr int NT   = COLS / 16;    // 4 or 2 n-tiles per wave
    constexpr int KC   = NN / SPLITK;  // K-chunk depth

    const int sup   = blockIdx.z;
    const int kc    = blockIdx.y;
    const int wave  = threadIdx.x >> 5;
    const int lane  = threadIdx.x & 31;
    const int l15   = lane & 15;
    const int mBase = blockIdx.x * 128 + wave * 16;

    v8f acc[NT];
#pragma unroll
    for (int t = 0; t < NT; ++t) acc[t] = (v8f){};

    const int kb = (lane < 16) ? 0 : 16;
    const u16* Arow = supB + (size_t)sup * NN * NN + (size_t)(mBase + l15) * NN + kc * KC;
    const u16* Bbase = xpk + (size_t)l15 * NN + kc * KC + ((lane < 16) ? 0 : 16);

    for (int kk = 0; kk < KC; kk += 32) {
        Frag a;
        const u16* ap = Arow + kk + kb;
        a.q[0] = *reinterpret_cast<const U4*>(ap);
        a.q[1] = *reinterpret_cast<const U4*>(ap + 8);
#pragma unroll
        for (int t = 0; t < NT; ++t) {
            Frag b;
            const u16* bp = Bbase + (size_t)t * 16 * NN + kk;
            b.q[0] = *reinterpret_cast<const U4*>(bp);
            b.q[1] = *reinterpret_cast<const U4*>(bp + 8);
            acc[t] = wmma_bf16(a, b, acc[t]);
        }
    }

    // write f32 partial x_cat into this K-chunk's slice (cols sup*F .. sup*F+F)
    const size_t ROWS = (size_t)BB * NN;
    float* base = xpart + (size_t)kc * ROWS * 4;
#pragma unroll
    for (int t = 0; t < NT; ++t) {
        int c  = t * 16 + l15;
        int bb = c / F, fi = c % F;
#pragma unroll
        for (int v = 0; v < 8; ++v) {
            int m = mBase + v + ((lane < 16) ? 0 : 8);
            size_t row = (size_t)bb * NN + m;
            base[row * 4 + sup * F + fi] = acc[t][v];
        }
    }
}

// ---------------- gates GEMM: [z|r] = sigmoid(A1 @ Wzr + b); pack r*h into A2 ----------------
__global__ __launch_bounds__(256) void gates_kernel(
    const u16* __restrict__ A1,        // [BB*NN][KPAD]
    const u16* __restrict__ Wzr,       // [128][KPAD] col-major packed (z cols 0..63, r cols 64..127)
    const float* __restrict__ bz, const float* __restrict__ br,
    const float* __restrict__ hbuf,    // [BB*NN][HH] f32
    float* __restrict__ zbuf,          // [BB*NN][HH] f32
    u16* __restrict__ A2,
    int hoff, int zeroTail)
{
    const int mt   = blockIdx.x * 8 + (threadIdx.x >> 5);
    const int lane = threadIdx.x & 31;
    const int l15  = lane & 15;
    const size_t rowBase = (size_t)mt * 16;

    v8f acc[8];
#pragma unroll
    for (int t = 0; t < 8; ++t) acc[t] = (v8f){};

    const size_t mA = rowBase + l15;
    const int kb = (lane < 16) ? 0 : 16;

#pragma unroll
    for (int k0 = 0; k0 < KPAD; k0 += 32) {
        Frag a;
        const u16* ap = A1 + mA * KPAD + k0 + kb;
        a.q[0] = *reinterpret_cast<const U4*>(ap);
        a.q[1] = *reinterpret_cast<const U4*>(ap + 8);
#pragma unroll
        for (int t = 0; t < 8; ++t) {
            Frag b;
            const u16* bp = Wzr + (size_t)(t * 16 + l15) * KPAD + k0 + ((lane < 16) ? 0 : 16);
            b.q[0] = *reinterpret_cast<const U4*>(bp);
            b.q[1] = *reinterpret_cast<const U4*>(bp + 8);
            acc[t] = wmma_bf16(a, b, acc[t]);
        }
    }

#pragma unroll
    for (int t = 0; t < 8; ++t) {
        int n = t * 16 + l15;                      // 0..127
        float bias = (n < HH) ? bz[n] : br[n - HH];
#pragma unroll
        for (int v = 0; v < 8; ++v) {
            size_t row = rowBase + v + ((lane < 16) ? 0 : 8);
            float g = 1.f / (1.f + expf(-(acc[t][v] + bias)));
            if (n < HH) {
                zbuf[row * HH + n] = g;
            } else {
                float rh = g * hbuf[row * HH + (n - HH)];
                A2[row * KPAD + hoff + (n - HH)] = f2bf(rh);
            }
        }
    }
    if (zeroTail && l15 < 2) {                     // decoder: cols 66,67 must be zero
#pragma unroll
        for (int v = 0; v < 8; ++v) {
            size_t row = rowBase + v + ((lane < 16) ? 0 : 8);
            A2[row * KPAD + 66 + l15] = 0;
        }
    }
}

// ---------------- candidate GEMM + GRU blend + LayerNorm (+ fused fc output) ----------------
__global__ __launch_bounds__(256) void cand_kernel(
    const u16* __restrict__ A2,
    const u16* __restrict__ WhT,       // [64][KPAD] col-major packed
    const float* __restrict__ bh,
    const float* __restrict__ gW, const float* __restrict__ betaW,
    const float* __restrict__ zbuf,
    float* __restrict__ hbuf,          // in/out (rows exclusive per wave)
    u16* __restrict__ A1,
    int hoffNext, int zeroTail,
    const float* __restrict__ fcW, const float* __restrict__ fcb,
    float* __restrict__ ybuf, float* __restrict__ dout,
    int doOut, int tdec)
{
    const int mt   = blockIdx.x * 8 + (threadIdx.x >> 5);
    const int lane = threadIdx.x & 31;
    const int l15  = lane & 15;
    const size_t rowBase = (size_t)mt * 16;

    v8f acc[4];
#pragma unroll
    for (int t = 0; t < 4; ++t) acc[t] = (v8f){};

    const size_t mA = rowBase + l15;
    const int kb = (lane < 16) ? 0 : 16;

#pragma unroll
    for (int k0 = 0; k0 < KPAD; k0 += 32) {
        Frag a;
        const u16* ap = A2 + mA * KPAD + k0 + kb;
        a.q[0] = *reinterpret_cast<const U4*>(ap);
        a.q[1] = *reinterpret_cast<const U4*>(ap + 8);
#pragma unroll
        for (int t = 0; t < 4; ++t) {
            Frag b;
            const u16* bp = WhT + (size_t)(t * 16 + l15) * KPAD + k0 + ((lane < 16) ? 0 : 16);
            b.q[0] = *reinterpret_cast<const U4*>(bp);
            b.q[1] = *reinterpret_cast<const U4*>(bp + 8);
            acc[t] = wmma_bf16(a, b, acc[t]);
        }
    }

    // h_tilde = tanh(pre + bh); h_new = (1-z)h + z*h_tilde   (stored back into acc)
#pragma unroll
    for (int t = 0; t < 4; ++t) {
        int n = t * 16 + l15;
#pragma unroll
        for (int v = 0; v < 8; ++v) {
            size_t row = rowBase + v + ((lane < 16) ? 0 : 8);
            float htld = tanhf(acc[t][v] + bh[n]);
            float z = zbuf[row * HH + n];
            float h = hbuf[row * HH + n];
            acc[t][v] = (1.f - z) * h + z * htld;
        }
    }

    // per-row LayerNorm across 64 cols: reduce over 4 tiles + 16 lanes of the half-wave
#pragma unroll
    for (int v = 0; v < 8; ++v) {
        float s = 0.f, ss = 0.f;
#pragma unroll
        for (int t = 0; t < 4; ++t) { float x = acc[t][v]; s += x; ss += x * x; }
#pragma unroll
        for (int o = 1; o < 16; o <<= 1) { s += __shfl_xor(s, o); ss += __shfl_xor(ss, o); }
        float mu  = s  * (1.f / 64.f);
        float var = ss * (1.f / 64.f) - mu * mu;
        float rs  = rsqrtf(var + 1e-5f);
#pragma unroll
        for (int t = 0; t < 4; ++t) {
            int n = t * 16 + l15;
            acc[t][v] = (acc[t][v] - mu) * rs * gW[n] + betaW[n];
        }
    }

    // write h (f32) and bf16-packed h into A1 at next step's offset
#pragma unroll
    for (int t = 0; t < 4; ++t) {
        int n = t * 16 + l15;
#pragma unroll
        for (int v = 0; v < 8; ++v) {
            size_t row = rowBase + v + ((lane < 16) ? 0 : 8);
            float x = acc[t][v];
            hbuf[row * HH + n] = x;
            A1[row * KPAD + hoffNext + n] = f2bf(x);
        }
    }
    if (zeroTail && l15 < 2) {
#pragma unroll
        for (int v = 0; v < 8; ++v) {
            size_t row = rowBase + v + ((lane < 16) ? 0 : 8);
            A1[row * KPAD + 66 + l15] = 0;
        }
    }

    // decoder: y = h_ln @ fc_W + fc_b -> ybuf (next diffusion input) + d_out
    if (doOut) {
#pragma unroll
        for (int v = 0; v < 8; ++v) {
            float p = 0.f;
#pragma unroll
            for (int t = 0; t < 4; ++t) p += acc[t][v] * fcW[t * 16 + l15];
#pragma unroll
            for (int o = 1; o < 16; o <<= 1) p += __shfl_xor(p, o);
            if (l15 == 0) {
                size_t row = rowBase + v + ((lane < 16) ? 0 : 8);
                float y = p + fcb[0];
                ybuf[row] = y;
                size_t b = row / NN, node = row % NN;
                dout[(b * HOR + (size_t)tdec) * NN + node] = y;
            }
        }
    }
}

// ---------------- host orchestration ----------------
extern "C" void kernel_launch(void* const* d_in, const int* in_sizes, int n_in,
                              void* d_out, int out_size, void* d_ws, size_t ws_size,
                              hipStream_t stream) {
    (void)in_sizes; (void)n_in; (void)out_size; (void)ws_size;

    const float* X      = (const float*)d_in[0];   // [B,T,N,2]
    const float* sup    = (const float*)d_in[1];   // [2,N,N]
    const float* encWz  = (const float*)d_in[2];
    const float* encbz  = (const float*)d_in[3];
    const float* encWr  = (const float*)d_in[4];
    const float* encbr  = (const float*)d_in[5];
    const float* encWh  = (const float*)d_in[6];
    const float* encbh  = (const float*)d_in[7];
    const float* encg   = (const float*)d_in[8];
    const float* encbe  = (const float*)d_in[9];
    const float* decWz  = (const float*)d_in[10];
    const float* decbz  = (const float*)d_in[11];
    const float* decWr  = (const float*)d_in[12];
    const float* decbr  = (const float*)d_in[13];
    const float* decWh  = (const float*)d_in[14];
    const float* decbh  = (const float*)d_in[15];
    const float* decg   = (const float*)d_in[16];
    const float* decbe  = (const float*)d_in[17];
    const float* fcW    = (const float*)d_in[18];
    const float* fcb    = (const float*)d_in[19];
    float* dout = (float*)d_out;

    // workspace carve-up (256B aligned)
    char* w = (char*)d_ws;
    size_t off = 0;
    auto carve = [&](size_t bytes) -> char* {
        char* p = w + off;
        off += (bytes + 255) & ~(size_t)255;
        return p;
    };
    const size_t ROWS = (size_t)BB * NN;
    u16*   supB  = (u16*)  carve((size_t)KDIF * NN * NN * sizeof(u16));
    u16*   A1    = (u16*)  carve(ROWS * KPAD * sizeof(u16));
    u16*   A2    = (u16*)  carve(ROWS * KPAD * sizeof(u16));
    float* hbuf  = (float*)carve(ROWS * HH * sizeof(float));
    float* zbuf  = (float*)carve(ROWS * HH * sizeof(float));
    float* ybuf  = (float*)carve(ROWS * sizeof(float));
    u16*   xpk   = (u16*)  carve((size_t)BB * FIN * NN * sizeof(u16));
    float* xpart = (float*)carve((size_t)SPLITK * ROWS * 4 * sizeof(float));
    u16*   wzrE  = (u16*)  carve(128 * KPAD * sizeof(u16));
    u16*   wzrD  = (u16*)  carve(128 * KPAD * sizeof(u16));
    u16*   whE   = (u16*)  carve(64 * KPAD * sizeof(u16));
    u16*   whD   = (u16*)  carve(64 * KPAD * sizeof(u16));

    // ---- prep (re-run every call: deterministic) ----
    bf16cast_kernel<<<8192, 256, 0, stream>>>(sup, supB, (size_t)KDIF * NN * NN);
    pack_kernel<<<(128 * KPAD + 255) / 256, 256, 0, stream>>>(encWz, encWr, wzrE, KDIF * FIN + HH, 128);
    pack_kernel<<<(128 * KPAD + 255) / 256, 256, 0, stream>>>(decWz, decWr, wzrD, KDIF * 1 + HH, 128);
    pack_kernel<<<(64 * KPAD + 255) / 256, 256, 0, stream>>>(encWh, encWh, whE, KDIF * FIN + HH, 64);
    pack_kernel<<<(64 * KPAD + 255) / 256, 256, 0, stream>>>(decWh, decWh, whD, KDIF * 1 + HH, 64);
    zero_kernel<<<4096, 256, 0, stream>>>((u32*)A1,   ROWS * KPAD * sizeof(u16) / 4);
    zero_kernel<<<4096, 256, 0, stream>>>((u32*)A2,   ROWS * KPAD * sizeof(u16) / 4);
    zero_kernel<<<4096, 256, 0, stream>>>((u32*)hbuf, ROWS * HH);
    zero_kernel<<<1024, 256, 0, stream>>>((u32*)ybuf, ROWS);

    const dim3 dgrid(NN / 128, SPLITK, KDIF);
    const int  gblocks = (int)(ROWS / 16 / 8);   // 1024 workgroups, 8 M-tiles each

    // ---- encoder: 12 steps ----
    for (int t = 0; t < TT; ++t) {
        const int colsE = BB * FIN;
        xpack_kernel<<<(colsE * NN + 255) / 256, 256, 0, stream>>>(
            X + (size_t)t * NN * FIN, (long)TT * NN * FIN, (long)FIN, FIN, xpk, colsE);
        diffusion_kernel<FIN><<<dgrid, 256, 0, stream>>>(supB, xpk, xpart);
        xmerge_kernel<<<(int)((ROWS * KDIF * FIN + 255) / 256), 256, 0, stream>>>(
            xpart, A1, A2, KDIF * FIN);
        gates_kernel<<<gblocks, 256, 0, stream>>>(
            A1, wzrE, encbz, encbr, hbuf, zbuf, A2, /*hoff=*/KDIF * FIN, /*zeroTail=*/0);
        int last = (t == TT - 1);
        cand_kernel<<<gblocks, 256, 0, stream>>>(
            A2, whE, encbh, encg, encbe, zbuf, hbuf, A1,
            /*hoffNext=*/last ? KDIF * 1 : KDIF * FIN, /*zeroTail=*/last,
            fcW, fcb, ybuf, dout, /*doOut=*/0, /*tdec=*/0);
    }

    // ---- decoder: 12 steps ----
    for (int s = 0; s < HOR; ++s) {
        const int colsD = BB;
        xpack_kernel<<<(colsD * NN + 255) / 256, 256, 0, stream>>>(
            ybuf, (long)NN, (long)1, 1, xpk, colsD);
        diffusion_kernel<1><<<dgrid, 256, 0, stream>>>(supB, xpk, xpart);
        xmerge_kernel<<<(int)((ROWS * KDIF + 255) / 256), 256, 0, stream>>>(
            xpart, A1, A2, KDIF);
        gates_kernel<<<gblocks, 256, 0, stream>>>(
            A1, wzrD, decbz, decbr, hbuf, zbuf, A2, /*hoff=*/KDIF * 1, /*zeroTail=*/1);
        cand_kernel<<<gblocks, 256, 0, stream>>>(
            A2, whD, decbh, decg, decbe, zbuf, hbuf, A1,
            /*hoffNext=*/KDIF * 1, /*zeroTail=*/1,
            fcW, fcb, ybuf, dout, /*doOut=*/1, /*tdec=*/s);
    }
}
